// GCN_31576599560908
// MI455X (gfx1250) — matrix operand order repaired and checked by reading, not compiled, not run on hardware
//
#include <hip/hip_runtime.h>
#include <hip/hip_bf16.h>
#include <cstdint>

// ---------------------------------------------------------------------------
// Types for CDNA5 WMMA (wave32): A/B = 16 bf16 (8 VGPRs), C/D = 8 f32.
// ---------------------------------------------------------------------------
typedef __bf16        v16bf __attribute__((ext_vector_type(16)));
typedef __bf16        v2bf  __attribute__((ext_vector_type(2)));
typedef float         v8f   __attribute__((ext_vector_type(8)));
typedef unsigned int  v8u   __attribute__((ext_vector_type(8)));

union BF16Frag { v8u u; v16bf h; };

#define EMB 256          // EMBED == IN_CH == 256
#define N_NODES_C 100000
#define N_EDGES_C 300000
#define LDA_PAD 17       // 64x32 A tile, padded dword stride (bank-conflict free)
#define LDA_LP 257       // link-pred A tile padded dword stride

// f32 -> packed bf16 pair (round to nearest even) ----------------------------
#if __has_builtin(__builtin_amdgcn_cvt_pk_bf16_f32)
__device__ __forceinline__ uint32_t pack2bf(float lo, float hi) {
  union { v2bf b; uint32_t u; } c;
  c.b = __builtin_amdgcn_cvt_pk_bf16_f32(lo, hi);   // v_cvt_pk_bf16_f32
  return c.u;
}
#else
__device__ __forceinline__ uint32_t f2bf(float f) {
  uint32_t u = __float_as_uint(f);
  uint32_t r = ((u >> 16) & 1u) + 0x7FFFu;
  return (u + r) >> 16;
}
__device__ __forceinline__ uint32_t pack2bf(float lo, float hi) {
  return f2bf(lo) | (f2bf(hi) << 16);
}
#endif

// ---------------------------------------------------------------------------
// Small utility kernels
// ---------------------------------------------------------------------------
__global__ void k_zero(float* __restrict__ p, int n) {
  int i = blockIdx.x * blockDim.x + threadIdx.x;
  if (i < n) p[i] = 0.0f;
}
__global__ void k_init_deg(float* __restrict__ deg, int n) {
  int i = blockIdx.x * blockDim.x + threadIdx.x;
  if (i < n) deg[i] = 1.0f;                       // self loop
}
__global__ void k_deg_edges(const int* __restrict__ dst, float* __restrict__ deg, int ne) {
  int e = blockIdx.x * blockDim.x + threadIdx.x;
  if (e < ne) atomicAdd(&deg[dst[e]], 1.0f);
}
__global__ void k_dinv(float* __restrict__ deg, int n) {
  int i = blockIdx.x * blockDim.x + threadIdx.x;
  if (i < n) deg[i] = rsqrtf(deg[i]);             // deg >= 1 always
}

// ---------------------------------------------------------------------------
// Preshuffle f32 weight W[K][N] (row-major) into bf16 B-fragment layout.
//   n = t*16 + (l&15);  k = s*32 + (l>=16 ? 16 : 0) + 2v ; even K in [15:0]
// Flat dword index: ((t*ksteps + s)*32 + lane)*8 + v
// ---------------------------------------------------------------------------
__global__ void k_preshuffle(const float* __restrict__ W, uint32_t* __restrict__ out,
                             int K, int N) {
  int total = (K >> 1) * N;
  int tid = blockIdx.x * blockDim.x + threadIdx.x;
  if (tid >= total) return;
  int ksteps = K >> 5;
  int v    = tid & 7;
  int lane = (tid >> 3) & 31;
  int ts   = tid >> 8;
  int s    = ts % ksteps;
  int t    = ts / ksteps;
  int n    = t * 16 + (lane & 15);
  int k    = s * 32 + ((lane >= 16) ? 16 : 0) + 2 * v;
  out[tid] = pack2bf(W[(size_t)k * N + n], W[(size_t)(k + 1) * N + n]);
}

// ---------------------------------------------------------------------------
// A-tile loader: 64 rows x 32 K, f32 -> packed bf16 dwords, BN affine + ReLU
// fused at compile time. `full` is a block-uniform fast path; dk (and hence
// the BN params) is invariant across the 4 fill iterations.
// ---------------------------------------------------------------------------
template <bool BN>
__device__ __forceinline__ void load_a_tile(
    uint32_t* __restrict__ buf, const float* __restrict__ X,
    int mBase, int k0, int M, bool full,
    const float* __restrict__ bnscale, const float* __restrict__ bnshift, int tid) {
  const int dk   = tid & 15;
  const int row0 = tid >> 4;
  const int k    = k0 + dk * 2;
  float sc0 = 0.f, sh0 = 0.f, sc1 = 0.f, sh1 = 0.f;
  if (BN) {
    sc0 = bnscale[k];     sh0 = bnshift[k];
    sc1 = bnscale[k + 1]; sh1 = bnshift[k + 1];
  }
  const float* base = X + (size_t)(mBase + row0) * EMB + k;
  if (full) {
#pragma unroll
    for (int i = 0; i < 4; ++i) {
      float2 v2 = *reinterpret_cast<const float2*>(base + (size_t)i * 16 * EMB);
      float a0 = v2.x, a1 = v2.y;
      if (BN) {
        a0 = fmaxf(a0 * sc0 + sh0, 0.0f);
        a1 = fmaxf(a1 * sc1 + sh1, 0.0f);
      }
      buf[(row0 + i * 16) * LDA_PAD + dk] = pack2bf(a0, a1);
    }
  } else {
#pragma unroll
    for (int i = 0; i < 4; ++i) {
      uint32_t pk = 0u;
      if (mBase + row0 + i * 16 < M) {
        float2 v2 = *reinterpret_cast<const float2*>(base + (size_t)i * 16 * EMB);
        float a0 = v2.x, a1 = v2.y;
        if (BN) {
          a0 = fmaxf(a0 * sc0 + sh0, 0.0f);
          a1 = fmaxf(a1 * sc1 + sh1, 0.0f);
        }
        pk = pack2bf(a0, a1);
      }
      buf[(row0 + i * 16) * LDA_PAD + dk] = pk;
    }
  }
}

// ---------------------------------------------------------------------------
// Node-feature GEMM:  Y[M,256] = act(X)[M,256] @ W[256,256]
// 64x64 block tile, 8 waves = 4 m-tiles x 2 n-pairs; each wave computes two
// 16x16 tiles reusing its A fragment. LDS A tile double-buffered so global
// fills of step s+1 overlap the WMMAs of step s. grid = (4, ceil(M/64)).
// ---------------------------------------------------------------------------
template <bool BN>
__global__ __launch_bounds__(256) void k_gemm_nodes(
    const float* __restrict__ X, const uint32_t* __restrict__ Bsh,
    float* __restrict__ Y, int M,
    const float* __restrict__ bnscale, const float* __restrict__ bnshift) {
  __shared__ uint32_t sA[2][64 * LDA_PAD];

  const int tid  = threadIdx.x;
  const int wave = tid >> 5;
  const int lane = tid & 31;
  const int mBase = blockIdx.y * 64;
  const int nBlk  = blockIdx.x * 64;              // 4 n-tiles per block
  const int mt = wave & 3;                        // 0..3
  const int np = wave >> 2;                       // 0..1 (n-pair)
  const int tA = (nBlk >> 4) + np * 2;            // global n-tile of acc0
  const int tB = tA + 1;                          // global n-tile of acc1
  const bool full = (mBase + 64) <= M;

  v8f acc0 = {}, acc1 = {};
  const int arow  = mt * 16 + (lane & 15);
  const int dbase = (lane >= 16) ? 8 : 0;

  load_a_tile<BN>(sA[0], X, mBase, 0, M, full, bnscale, bnshift, tid);

#pragma unroll
  for (int s = 0; s < 8; ++s) {
    __syncthreads();                              // buf[s&1] ready
    if (s + 1 < 8)                                // prefetch next step
      load_a_tile<BN>(sA[(s + 1) & 1], X, mBase, (s + 1) * 32, M, full,
                      bnscale, bnshift, tid);

    const uint32_t* cur = sA[s & 1];
    BF16Frag a, b0, b1;
#pragma unroll
    for (int v = 0; v < 8; ++v) {
      int d = v + dbase;                          // contiguous k = 16h + 2v
      a.u[v] = cur[arow * LDA_PAD + d];
    }
    b0.u = *reinterpret_cast<const v8u*>(Bsh + (((size_t)tA * 8 + s) * 32 + lane) * 8);
    b1.u = *reinterpret_cast<const v8u*>(Bsh + (((size_t)tB * 8 + s) * 32 + lane) * 8);
    acc0 = __builtin_amdgcn_wmma_f32_16x16x32_bf16(false, a.h, false, b0.h,
                                                   (short)0, acc0, false, false);
    acc1 = __builtin_amdgcn_wmma_f32_16x16x32_bf16(false, a.h, false, b1.h,
                                                   (short)0, acc1, false, false);
  }

  // C layout: lane<16 -> M=r, N=lane ; lane>=16 -> M=8+r, N=lane-16
  const int crow  = mt * 16 + ((lane >= 16) ? 8 : 0);
  const int ccolA = tA * 16 + (lane & 15);
  const int ccolB = tB * 16 + (lane & 15);
#pragma unroll
  for (int r = 0; r < 8; ++r) {
    int grow = mBase + crow + r;
    if (full || grow < M) {
      Y[(size_t)grow * EMB + ccolA] = acc0[r];
      Y[(size_t)grow * EMB + ccolB] = acc1[r];
    }
  }
}

// ---------------------------------------------------------------------------
// Aggregation: out = bias + dinv^2 * xw   (self loops, no atomics)
// ---------------------------------------------------------------------------
__global__ void k_agg_self(const float* __restrict__ xw, const float* __restrict__ dinv,
                           const float* __restrict__ bias, float* __restrict__ out, int M) {
  size_t idx = (size_t)blockIdx.x * blockDim.x + threadIdx.x;
  if (idx >= (size_t)M * EMB) return;
  int i = (int)(idx >> 8);
  int f = (int)(idx & 255);
  float di = dinv[i];
  out[idx] = bias[f] + di * di * xw[idx];
}

// Edge scatter: out[dst] += dinv[src]*dinv[dst] * xw[src]  (f32 global atomics)
__global__ __launch_bounds__(256) void k_agg_edges(
    const float* __restrict__ xw, const int* __restrict__ src,
    const int* __restrict__ dst, const float* __restrict__ dinv,
    float* __restrict__ out, int ne) {
  int e = blockIdx.x;
  int f = threadIdx.x;
  if (e >= ne) return;
  int s = src[e], d = dst[e];
  float norm = dinv[s] * dinv[d];
  atomicAdd(&out[(size_t)d * EMB + f], norm * xw[(size_t)s * EMB + f]);
}

// ---------------------------------------------------------------------------
// BatchNorm: per-channel partial sums + finalize into scale/shift.
// ---------------------------------------------------------------------------
__global__ __launch_bounds__(256) void k_bn_stats(
    const float* __restrict__ x, float* __restrict__ sums,
    float* __restrict__ sumsq, int M) {
  int f  = threadIdx.x;
  int r0 = blockIdx.x * 256;
  int r1 = r0 + 256; if (r1 > M) r1 = M;
  float s = 0.f, s2 = 0.f;
  for (int r = r0; r < r1; ++r) {
    float v = x[(size_t)r * EMB + f];
    s += v; s2 += v * v;
  }
  atomicAdd(&sums[f], s);
  atomicAdd(&sumsq[f], s2);
}

__global__ void k_bn_finalize(const float* __restrict__ sums, const float* __restrict__ sumsq,
                              const float* __restrict__ g, const float* __restrict__ be,
                              float* __restrict__ scale, float* __restrict__ shift,
                              float invM) {
  int f = threadIdx.x;
  float mu  = sums[f] * invM;
  float var = sumsq[f] * invM - mu * mu;
  float sc  = g[f] * rsqrtf(var + 1e-5f);
  scale[f] = sc;
  shift[f] = be[f] - mu * sc;
}

// ---------------------------------------------------------------------------
// Fused link predictor: 32 edges/block.
//   h = relu(concat(z[a],z[b]) @ fc1_w + fc1_b); out = sigmoid(h @ fc2_w + fc2_b)
// A (32x512) gathered+converted into LDS (padded stride); the ia/ib pointer
// select is hoisted out of the gather loop (wave-uniform, branch-free).
// 8 waves, each owns n-tiles {w, w+8} and both 16-row tiles -> 4 accumulators,
// 4 WMMAs per k-step reusing 2 B fragments. fc2 dot fused via LDS f32 atomics.
// ---------------------------------------------------------------------------
__global__ __launch_bounds__(256) void k_linkpred(
    const float* __restrict__ Z, const int* __restrict__ ia, const int* __restrict__ ib,
    const uint32_t* __restrict__ Bsh, const float* __restrict__ fc1b,
    const float* __restrict__ fc2w, const float* __restrict__ fc2b,
    float* __restrict__ out, int ne) {
  __shared__ uint32_t sA[32 * LDA_LP];            // 32 rows x 512 K packed (~33 KB)
  __shared__ float hsum[32];

  const int tid  = threadIdx.x;
  const int lane = tid & 31;
  const int wave = tid >> 5;
  const int eBase = blockIdx.x * 32;

  // gather: iteration i fills row i; threads 0-127 (waves 0-3) supply z[ia],
  // threads 128-255 supply z[ib]; select hoisted -> no branches in loop.
  const int* __restrict__ idxp = (tid < 128) ? ia : ib;
  const int kloc = (tid & 127) * 2;
#pragma unroll 4
  for (int i = 0; i < 32; ++i) {
    int node = idxp[eBase + i];
    float2 v2 = *reinterpret_cast<const float2*>(Z + (size_t)node * EMB + kloc);
    sA[i * LDA_LP + tid] = pack2bf(v2.x, v2.y);
  }
  if (tid < 32) hsum[tid] = fc2b[0];
  __syncthreads();

  v8f acc00 = {}, acc01 = {}, acc10 = {}, acc11 = {};
  const int t0 = wave, t1 = wave + 8;
  const int arow0 = (lane & 15);                  // rows 0..15
  const int arow1 = 16 + (lane & 15);             // rows 16..31
  const int dbase = (lane >= 16) ? 8 : 0;

#pragma unroll
  for (int s = 0; s < 16; ++s) {
    BF16Frag a0, a1, b0, b1;
#pragma unroll
    for (int v = 0; v < 8; ++v) {
      int d = s * 16 + v + dbase;
      a0.u[v] = sA[arow0 * LDA_LP + d];
      a1.u[v] = sA[arow1 * LDA_LP + d];
    }
    b0.u = *reinterpret_cast<const v8u*>(Bsh + (((size_t)t0 * 16 + s) * 32 + lane) * 8);
    b1.u = *reinterpret_cast<const v8u*>(Bsh + (((size_t)t1 * 16 + s) * 32 + lane) * 8);
    acc00 = __builtin_amdgcn_wmma_f32_16x16x32_bf16(false, a0.h, false, b0.h,
                                                    (short)0, acc00, false, false);
    acc01 = __builtin_amdgcn_wmma_f32_16x16x32_bf16(false, a0.h, false, b1.h,
                                                    (short)0, acc01, false, false);
    acc10 = __builtin_amdgcn_wmma_f32_16x16x32_bf16(false, a1.h, false, b0.h,
                                                    (short)0, acc10, false, false);
    acc11 = __builtin_amdgcn_wmma_f32_16x16x32_bf16(false, a1.h, false, b1.h,
                                                    (short)0, acc11, false, false);
  }

  const int col0 = t0 * 16 + (lane & 15);
  const int col1 = t1 * 16 + (lane & 15);
  const float w0 = fc2w[col0], w1 = fc2w[col1];
  const float fb0 = fc1b[col0], fb1 = fc1b[col1];
  const int rbase = (lane >= 16) ? 8 : 0;
#pragma unroll
  for (int r = 0; r < 8; ++r) {
    float p0 = fmaxf(acc00[r] + fb0, 0.0f) * w0 + fmaxf(acc01[r] + fb1, 0.0f) * w1;
    float p1 = fmaxf(acc10[r] + fb0, 0.0f) * w0 + fmaxf(acc11[r] + fb1, 0.0f) * w1;
    atomicAdd(&hsum[rbase + r], p0);              // ds_add_f32
    atomicAdd(&hsum[16 + rbase + r], p1);
  }
  __syncthreads();
  if (tid < 32) {
    int e = eBase + tid;
    if (e < ne) out[e] = 1.0f / (1.0f + __expf(-hsum[tid]));
  }
}

// ---------------------------------------------------------------------------
// Host-side launcher
// ---------------------------------------------------------------------------
static inline size_t alignup(size_t x) { return (x + 1023) & ~(size_t)1023; }

extern "C" void kernel_launch(void* const* d_in, const int* in_sizes, int n_in,
                              void* d_out, int out_size, void* d_ws, size_t ws_size,
                              hipStream_t stream) {
  (void)in_sizes; (void)n_in; (void)out_size; (void)ws_size;
  const float* node_feat = (const float*)d_in[0];
  const int*   src  = (const int*)d_in[1];
  const int*   dst  = (const int*)d_in[2];
  const int*   neg  = (const int*)d_in[3];
  const float* W1 = (const float*)d_in[4];  const float* b1 = (const float*)d_in[5];
  const float* W2 = (const float*)d_in[6];  const float* b2 = (const float*)d_in[7];
  const float* W3 = (const float*)d_in[8];  const float* b3 = (const float*)d_in[9];
  const float* g1 = (const float*)d_in[10]; const float* be1 = (const float*)d_in[11];
  const float* g2 = (const float*)d_in[12]; const float* be2 = (const float*)d_in[13];
  const float* fc1_w = (const float*)d_in[14]; const float* fc1_b = (const float*)d_in[15];
  const float* fc2_w = (const float*)d_in[16]; const float* fc2_b = (const float*)d_in[17];
  float* out = (float*)d_out;

  const int M = N_NODES_C, E = N_EDGES_C;
  const size_t NB = (size_t)M * EMB * sizeof(float);   // 102.4 MB

  char* ws = (char*)d_ws;
  size_t off = 0;
  float*    xw   = (float*)(ws + off);    off += alignup(NB);
  float*    act  = (float*)(ws + off);    off += alignup(NB);
  float*    dinv = (float*)(ws + off);    off += alignup((size_t)M * 4);
  float*    bns  = (float*)(ws + off);    off += alignup(EMB * 4);
  float*    bnsq = (float*)(ws + off);    off += alignup(EMB * 4);
  float*    bsc  = (float*)(ws + off);    off += alignup(EMB * 4);
  float*    bsh  = (float*)(ws + off);    off += alignup(EMB * 4);
  uint32_t* w1s  = (uint32_t*)(ws + off); off += alignup((size_t)EMB * EMB * 2);
  uint32_t* w2s  = (uint32_t*)(ws + off); off += alignup((size_t)EMB * EMB * 2);
  uint32_t* w3s  = (uint32_t*)(ws + off); off += alignup((size_t)EMB * EMB * 2);
  uint32_t* f1s  = (uint32_t*)(ws + off); off += alignup((size_t)2 * EMB * EMB * 2);

  // --- degrees / dinv (graph-invariant, reused by all 3 layers)
  k_init_deg<<<(M + 255) / 256, 256, 0, stream>>>(dinv, M);
  k_deg_edges<<<(E + 255) / 256, 256, 0, stream>>>(dst, dinv, E);
  k_dinv<<<(M + 255) / 256, 256, 0, stream>>>(dinv, M);

  // --- weight preshuffle to bf16 WMMA B-fragments
  k_preshuffle<<<(EMB * EMB / 2 + 255) / 256, 256, 0, stream>>>(W1, w1s, EMB, EMB);
  k_preshuffle<<<(EMB * EMB / 2 + 255) / 256, 256, 0, stream>>>(W2, w2s, EMB, EMB);
  k_preshuffle<<<(EMB * EMB / 2 + 255) / 256, 256, 0, stream>>>(W3, w3s, EMB, EMB);
  k_preshuffle<<<(2 * EMB * EMB / 2 + 255) / 256, 256, 0, stream>>>(fc1_w, f1s, 2 * EMB, EMB);

  const dim3 gGemm(4, (M + 63) / 64);
  const int  gElem = (int)(((size_t)M * EMB + 255) / 256);
  const float invM = 1.0f / (float)M;

  // --- layer 1: xw = node_feat @ W1 ; act = agg(xw) + b1
  k_gemm_nodes<false><<<gGemm, 256, 0, stream>>>(node_feat, w1s, xw, M, nullptr, nullptr);
  k_agg_self<<<gElem, 256, 0, stream>>>(xw, dinv, b1, act, M);
  k_agg_edges<<<E, 256, 0, stream>>>(xw, src, dst, dinv, act, E);
  // BN1 -> scale/shift
  k_zero<<<2, 256, 0, stream>>>(bns, EMB);
  k_zero<<<2, 256, 0, stream>>>(bnsq, EMB);
  k_bn_stats<<<(M + 255) / 256, 256, 0, stream>>>(act, bns, bnsq, M);
  k_bn_finalize<<<1, 256, 0, stream>>>(bns, bnsq, g1, be1, bsc, bsh, invM);

  // --- layer 2 (BN1+ReLU fused into A load)
  k_gemm_nodes<true><<<gGemm, 256, 0, stream>>>(act, w2s, xw, M, bsc, bsh);
  k_agg_self<<<gElem, 256, 0, stream>>>(xw, dinv, b2, act, M);
  k_agg_edges<<<E, 256, 0, stream>>>(xw, src, dst, dinv, act, E);
  // BN2
  k_zero<<<2, 256, 0, stream>>>(bns, EMB);
  k_zero<<<2, 256, 0, stream>>>(bnsq, EMB);
  k_bn_stats<<<(M + 255) / 256, 256, 0, stream>>>(act, bns, bnsq, M);
  k_bn_finalize<<<1, 256, 0, stream>>>(bns, bnsq, g2, be2, bsc, bsh, invM);

  // --- layer 3 -> z in `act`
  k_gemm_nodes<true><<<gGemm, 256, 0, stream>>>(act, w3s, xw, M, bsc, bsh);
  k_agg_self<<<gElem, 256, 0, stream>>>(xw, dinv, b3, act, M);
  k_agg_edges<<<E, 256, 0, stream>>>(xw, src, dst, dinv, act, E);

  // --- link predictor: pos then neg (outputs concatenated)
  k_linkpred<<<E / 32, 256, 0, stream>>>(act, src, dst, f1s, fc1_b, fc2_w, fc2_b, out, E);
  k_linkpred<<<E / 32, 256, 0, stream>>>(act, src, neg, f1s, fc1_b, fc2_w, fc2_b, out + E, E);
}
